// AttentionLayer_8203387536044
// MI455X (gfx1250) — matrix operand order repaired and checked by reading, not compiled, not run on hardware
//
#include <hip/hip_runtime.h>
#include <hip/hip_bf16.h>

// ---------------------------------------------------------------------------
// GAT layer for MI455X (gfx1250, wave32, WMMA).
//  k_cvt_x       : x f32 -> f16
//  k_wt          : W f32 [256,128] -> WT f16 [128,256]
//  k_gemm_h      : hT f16 [b][128][4096] = (x @ W)^T        (WMMA f16)
//  k_wh          : Wh1/Wh2 [b][4096]
//  k_colsum_part : 4-way-split partial column softmax denominators
//  k_colsum_fin  : csinv[b][j] = 1 / sum of partials
//  k_attn        : G = gelu(P @ h), P built on the fly, software-pipelined WMMA
//  k_ssq/k_norm  : deterministic column sum-of-squares reduction
//  k_finish      : out = G * ninv[b][d] + bias[d]
// ---------------------------------------------------------------------------

typedef __attribute__((ext_vector_type(16))) _Float16 v16h;
typedef __attribute__((ext_vector_type(8)))  float    v8f;

__device__ __forceinline__ float gelu_f(float x) {
  return 0.5f * x * (1.0f + erff(x * 0.70710678118654752f));
}

// A-operand (16x32 f16, M on lanes, K packed): lane<16 -> K {k0..k0+7, k0+16..k0+23},
// lane>=16 -> shifted by 8. Two 16B loads per lane.
__device__ __forceinline__ v16h load_tileA(const _Float16* __restrict__ base, int ld,
                                           int row0, int k0) {
  const int lane = threadIdx.x & 31;
  const int sub  = lane & 15;
  const int hi   = lane >> 4;
  const _Float16* p = base + (size_t)(row0 + sub) * ld + (k0 + hi * 8);
  union { float4 q[2]; v16h v; } u;
  u.q[0] = *(const float4*)(p);
  u.q[1] = *(const float4*)(p + 16);
  return u.v;
}

// B-operand (32x16 f16, N on lanes, K packed): lane<16 -> K {k0..k0+7, k0+16..k0+23},
// lane>=16 -> shifted by 8. Two 16B loads per lane.
__device__ __forceinline__ v16h load_tileB(const _Float16* __restrict__ base, int ld,
                                           int n0, int k0) {
  const int lane = threadIdx.x & 31;
  const int sub  = lane & 15;
  const int hi   = lane >> 4;
  const _Float16* p = base + (size_t)(n0 + sub) * ld + (k0 + hi * 8);
  union { float4 q[2]; v16h v; } u;
  u.q[0] = *(const float4*)(p);
  u.q[1] = *(const float4*)(p + 16);
  return u.v;
}

// Raw operands for one 8-wide j-group of the P tile.
struct RawG {
  int4   a0, a1;
  float4 w0, w1, c0, c1;
};

__device__ __forceinline__ RawG load_raw(const int* __restrict__ arow,
                                         const float* __restrict__ wv,
                                         const float* __restrict__ cv, int jb) {
  RawG r;
  r.a0 = *(const int4*)(arow + jb);
  r.a1 = *(const int4*)(arow + jb + 4);
  r.w0 = *(const float4*)(wv + jb);
  r.w1 = *(const float4*)(wv + jb + 4);
  r.c0 = *(const float4*)(cv + jb);
  r.c1 = *(const float4*)(cv + jb + 4);
  return r;
}

__device__ __forceinline__ void fill8(_Float16* dst, const RawG& r, float z) {
  const int   ai[8] = {r.a0.x, r.a0.y, r.a0.z, r.a0.w, r.a1.x, r.a1.y, r.a1.z, r.a1.w};
  const float wf[8] = {r.w0.x, r.w0.y, r.w0.z, r.w0.w, r.w1.x, r.w1.y, r.w1.z, r.w1.w};
  const float cf[8] = {r.c0.x, r.c0.y, r.c0.z, r.c0.w, r.c1.x, r.c1.y, r.c1.z, r.c1.w};
#pragma unroll
  for (int t = 0; t < 8; ++t) {
    float p = (ai[t] > 0) ? __expf(gelu_f(z + wf[t])) * cf[t] : 0.0f;
    dst[t] = (_Float16)p;
  }
}

// ------------------------------ prep kernels -------------------------------

__global__ void __launch_bounds__(256) k_cvt_x(const float* __restrict__ x,
                                               _Float16* __restrict__ xf16) {
  const int idx = blockIdx.x * 256 + threadIdx.x;   // each thread: 4 elements
  float4 v = ((const float4*)x)[idx];
  union { _Float16 h[4]; float2 q; } u;
  u.h[0] = (_Float16)v.x; u.h[1] = (_Float16)v.y;
  u.h[2] = (_Float16)v.z; u.h[3] = (_Float16)v.w;
  ((float2*)xf16)[idx] = u.q;
}

__global__ void __launch_bounds__(256) k_wt(const float* __restrict__ W,
                                            _Float16* __restrict__ WT) {
  const int idx = blockIdx.x * 256 + threadIdx.x;   // 32768 = 128*256
  const int d = idx >> 8;
  const int k = idx & 255;
  WT[idx] = (_Float16)W[k * 128 + d];
}

// ------------------------- h = x @ W  (WMMA GEMM) --------------------------
__global__ void __launch_bounds__(128) k_gemm_h(const _Float16* __restrict__ xf16,
                                                const _Float16* __restrict__ WT,
                                                _Float16* __restrict__ hT) {
  const int wid  = threadIdx.x >> 5;
  const int row0 = blockIdx.x * 64 + wid * 16;      // [0, 16384)
  const int lane = threadIdx.x & 31;
  const int sub  = lane & 15;
  const int hi   = lane >> 4;

  v8f acc[8];
#pragma unroll
  for (int ct = 0; ct < 8; ++ct) acc[ct] = {};

  for (int k0 = 0; k0 < 256; k0 += 32) {
    // Issue A + all 8 B-tile loads before any WMMA: one partial wait, 8 MMAs.
    v16h A = load_tileA(xf16, 256, row0, k0);
    v16h Bt[8];
#pragma unroll
    for (int ct = 0; ct < 8; ++ct) Bt[ct] = load_tileB(WT, 256, ct * 16, k0);
#pragma unroll
    for (int ct = 0; ct < 8; ++ct) {
      acc[ct] = __builtin_amdgcn_wmma_f32_16x16x32_f16(
          false, A, false, Bt[ct], (short)0, acc[ct], false, false);
    }
  }

  const int b  = row0 >> 12;
  const int il = row0 & 4095;
  _Float16* hTb = hT + (size_t)b * 128 * 4096;
#pragma unroll
  for (int ct = 0; ct < 8; ++ct) {
    const int d = ct * 16 + sub;                    // N index -> hT row
    union { _Float16 h8[8]; float4 q; } s;
#pragma unroll
    for (int r = 0; r < 8; ++r) s.h8[r] = (_Float16)acc[ct][r];  // M = hi*8 + r
    *(float4*)(hTb + (size_t)d * 4096 + il + hi * 8) = s.q;
  }
}

// ----------------------- Wh1/Wh2 (tiny projections) ------------------------
__global__ void __launch_bounds__(256) k_wh(const _Float16* __restrict__ hT,
                                            const float* __restrict__ a,
                                            float* __restrict__ Wh1,
                                            float* __restrict__ Wh2) {
  const int b = blockIdx.x >> 4;
  const int n = ((blockIdx.x & 15) << 8) + threadIdx.x;   // [0,4096)
  const _Float16* hTb = hT + (size_t)b * 128 * 4096;
  float s1 = 0.f, s2 = 0.f;
#pragma unroll 4
  for (int d = 0; d < 128; ++d) {
    float hv = (float)hTb[(size_t)d * 4096 + n];
    s1 = fmaf(hv, a[d], s1);
    s2 = fmaf(hv, a[128 + d], s2);
  }
  Wh1[b * 4096 + n] = s1;
  Wh2[b * 4096 + n] = s2;
}

// --------------------- column softmax denominators -------------------------
// softmax over axis=1 => normalize over i for each (b,j). No max-shift needed:
// unmasked values are gelu() outputs bounded far below exp-overflow, masked = 0.
// 4-way split over i for memory-level parallelism; deterministic combine below.
__global__ void __launch_bounds__(256) k_colsum_part(const int* __restrict__ adj,
                                                     const float* __restrict__ Wh1,
                                                     const float* __restrict__ Wh2,
                                                     float* __restrict__ part_cs) {
  const int blk = blockIdx.x;            // [b:2][ic:2][jc:4]
  const int b  = blk >> 6;
  const int ic = (blk >> 4) & 3;
  const int j  = ((blk & 15) << 8) + threadIdx.x;
  const float w = Wh2[b * 4096 + j];
  const int* adjb = adj + (size_t)b * 4096 * 4096 + (size_t)(ic * 1024) * 4096 + j;
  const float* z = Wh1 + b * 4096 + ic * 1024;            // uniform -> scalar loads
  float acc = 0.f;
#pragma unroll 8
  for (int i = 0; i < 1024; ++i) {
    int av = __builtin_nontemporal_load(adjb + (size_t)i * 4096);
    float e = __expf(gelu_f(z[i] + w));
    acc += (av > 0) ? e : 0.0f;
  }
  part_cs[((b * 4 + ic) << 12) + j] = acc;
}

__global__ void __launch_bounds__(256) k_colsum_fin(const float* __restrict__ part_cs,
                                                    float* __restrict__ csinv) {
  const int b = blockIdx.x >> 4;
  const int j = ((blockIdx.x & 15) << 8) + threadIdx.x;
  float s = 0.f;
#pragma unroll
  for (int ic = 0; ic < 4; ++ic) s += part_cs[((b * 4 + ic) << 12) + j];
  csinv[b * 4096 + j] = 1.0f / fmaxf(s, 1e-30f);
}

// ------------------- G = gelu(P @ h), P built on the fly -------------------
// Software-pipelined: per K-step, issue all 8 B-tile loads, build the A tile
// from raw operands prefetched on the previous iteration (VALU overlaps the
// loads), then issue the NEXT iteration's raw loads, then run 8 WMMAs. Since
// loads retire in order, the WMMA wait only drains the B tiles, keeping the
// adj stream in flight across the matrix ops.
__global__ void __launch_bounds__(128) k_attn(const int* __restrict__ adj,
                                              const float* __restrict__ Wh1,
                                              const float* __restrict__ Wh2,
                                              const float* __restrict__ csinv,
                                              const _Float16* __restrict__ hT,
                                              float* __restrict__ G) {
  const int wid  = threadIdx.x >> 5;
  const int row0 = blockIdx.x * 64 + wid * 16;
  const int lane = threadIdx.x & 31;
  const int sub  = lane & 15;
  const int hi   = lane >> 4;
  const int b    = row0 >> 12;
  const int il   = row0 & 4095;

  const int   i    = il + sub;                       // A-matrix: M = lane & 15
  const float z    = Wh1[b * 4096 + i];
  const int*  arow = adj + ((size_t)b * 4096 + i) * 4096;
  const float* wv  = Wh2 + b * 4096;
  const float* cv  = csinv + b * 4096;
  const _Float16* hTb = hT + (size_t)b * 128 * 4096;

  v8f acc[8];
#pragma unroll
  for (int ct = 0; ct < 8; ++ct) acc[ct] = {};

  // Prologue: raw operands for k0 = 0.
  RawG r0 = load_raw(arow, wv, cv, hi * 8);
  RawG r1 = load_raw(arow, wv, cv, 16 + hi * 8);

  for (int k0 = 0; k0 < 4096; k0 += 32) {
    // (1) issue B-tile loads for this K-step
    v16h Bt[8];
#pragma unroll
    for (int ct = 0; ct < 8; ++ct) Bt[ct] = load_tileB(hTb, 4096, ct * 16, k0);

    // (2) build A tile from prefetched raw operands (VALU hides load latency)
    union { _Float16 h16[16]; v16h v; } A;
    fill8(A.h16 + 0, r0, z);
    fill8(A.h16 + 8, r1, z);

    // (3) prefetch next K-step's raw operands (issued AFTER B loads, so the
    //     WMMA wait below leaves them outstanding); wrap keeps it in-bounds.
    const int kn = (k0 + 32) & 4095;
    r0 = load_raw(arow, wv, cv, kn + hi * 8);
    r1 = load_raw(arow, wv, cv, kn + 16 + hi * 8);

    // (4) 8 back-to-back WMMAs
#pragma unroll
    for (int ct = 0; ct < 8; ++ct) {
      acc[ct] = __builtin_amdgcn_wmma_f32_16x16x32_f16(
          false, A.v, false, Bt[ct], (short)0, acc[ct], false, false);
    }
  }

#pragma unroll
  for (int ct = 0; ct < 8; ++ct) {
    const int d = ct * 16 + sub;
#pragma unroll
    for (int r = 0; r < 8; ++r) {
      const int m = hi * 8 + r;                      // C layout: M = hi*8 + r
      G[((size_t)row0 + m) * 128 + d] = gelu_f(acc[ct][r]);
    }
  }
}

// -------------- L2 norm over node dim (deterministic 2-stage) --------------
__global__ void __launch_bounds__(128) k_ssq(const float* __restrict__ G,
                                             float* __restrict__ part) {
  const int b = blockIdx.x >> 4;
  const int c = blockIdx.x & 15;
  const int d = threadIdx.x;                         // [0,128)
  const float* p = G + ((size_t)b * 4096 + (size_t)c * 256) * 128 + d;
  float s = 0.f;
#pragma unroll 4
  for (int i = 0; i < 256; ++i) {
    float v = p[(size_t)i * 128];
    s = fmaf(v, v, s);
  }
  part[blockIdx.x * 128 + d] = s;
}

__global__ void __launch_bounds__(512) k_norm(const float* __restrict__ part,
                                              float* __restrict__ ninv) {
  const int t = threadIdx.x;                         // (b,d)
  const int b = t >> 7, d = t & 127;
  float s = 0.f;
#pragma unroll
  for (int c = 0; c < 16; ++c) s += part[(b * 16 + c) * 128 + d];
  ninv[t] = 1.0f / fmaxf(sqrtf(s), 1e-12f);
}

__global__ void __launch_bounds__(256) k_finish(float* __restrict__ out,
                                                const float* __restrict__ ninv,
                                                const float* __restrict__ bias) {
  const size_t idx  = (size_t)blockIdx.x * 256 + threadIdx.x;  // float4 index
  const size_t base = idx * 4;
  const int d0 = (int)(base & 127);
  const int b  = (int)(base >> 19);                  // 4096*128 = 2^19 per batch
  float4 v = ((const float4*)out)[idx];
  v.x = v.x * ninv[b * 128 + d0 + 0] + bias[d0 + 0];
  v.y = v.y * ninv[b * 128 + d0 + 1] + bias[d0 + 1];
  v.z = v.z * ninv[b * 128 + d0 + 2] + bias[d0 + 2];
  v.w = v.w * ninv[b * 128 + d0 + 3] + bias[d0 + 3];
  ((float4*)out)[idx] = v;
}

// ------------------------------- launcher ----------------------------------

extern "C" void kernel_launch(void* const* d_in, const int* in_sizes, int n_in,
                              void* d_out, int out_size, void* d_ws, size_t ws_size,
                              hipStream_t stream) {
  const float* x    = (const float*)d_in[0];   // [4,4096,256]
  const int*   adj  = (const int*)d_in[1];     // [4,4096,4096]
  const float* W    = (const float*)d_in[2];   // [256,128]
  const float* a    = (const float*)d_in[3];   // [256,1]
  const float* bias = (const float*)d_in[4];   // [128]
  float* out = (float*)d_out;                  // [4,4096,128]

  char* ws = (char*)d_ws;
  auto carve = [&](size_t bytes) -> char* {
    char* p = ws;
    ws += (bytes + 255) & ~(size_t)255;
    return p;
  };
  _Float16* xf16 = (_Float16*)carve((size_t)16384 * 256 * 2);    // 8.39 MB
  _Float16* WT   = (_Float16*)carve((size_t)128 * 256 * 2);      // 64 KB
  _Float16* hT   = (_Float16*)carve((size_t)4 * 128 * 4096 * 2); // 4.19 MB
  float* Wh1     = (float*)carve((size_t)16384 * 4);
  float* Wh2     = (float*)carve((size_t)16384 * 4);
  float* csinv   = (float*)carve((size_t)16384 * 4);
  float* part_cs = (float*)carve((size_t)16 * 4096 * 4);         // 256 KB
  float* part    = (float*)carve((size_t)64 * 128 * 4);
  float* ninv    = (float*)carve((size_t)512 * 4);

  k_cvt_x      <<<4096, 256, 0, stream>>>(x, xf16);
  k_wt         <<<128,  256, 0, stream>>>(W, WT);
  k_gemm_h     <<<256,  128, 0, stream>>>(xf16, WT, hT);
  k_wh         <<<64,   256, 0, stream>>>(hT, a, Wh1, Wh2);
  k_colsum_part<<<256,  256, 0, stream>>>(adj, Wh1, Wh2, part_cs);
  k_colsum_fin <<<64,   256, 0, stream>>>(part_cs, csinv);
  k_attn       <<<256,  128, 0, stream>>>(adj, Wh1, Wh2, csinv, hT, out);
  k_ssq        <<<64,   128, 0, stream>>>(out, part);
  k_norm       <<<1,    512, 0, stream>>>(part, ninv);
  k_finish     <<<2048, 256, 0, stream>>>(out, ninv, bias);
}
